// WitnessAttention_30021821399047
// MI455X (gfx1250) — matrix operand order repaired and checked by reading, not compiled, not run on hardware
//
#include <hip/hip_runtime.h>
#include <hip/hip_bf16.h>

// ---------------------------------------------------------------------------
// WitnessAttention for MI455X (gfx1250, wave32, WMMA).
//
// Pipeline:
//   1) topk_densify: per-token top-64 |x| selection (bit-binary-search on
//      abs-float keys), scatter to dense bf16 rows (qd, kd).
//   2) gemm_bf16_nt (V_WMMA_F32_16X16X32_BF16, f32 accum, double-buffered
//      GLOBAL_LOAD_ASYNC_TO_LDS_B128 staging):
//        scores = qd @ kd^T           -> f32, written into d_out attn region
//        vprojT = (v @ Wv^T)^T        -> bf16 (transposed so next GEMM is NT)
//        out1   = attn_bf @ vprojT^T  -> bf16
//        out    = out1 @ Wo^T         -> f32 into d_out
//   3) softmax_row: in-place f32 softmax on d_out attn + bf16 copy for GEMM.
//
// All GEMM dims are multiples of the 128x128x64 tile -> no boundary guards.
// ---------------------------------------------------------------------------

typedef __attribute__((ext_vector_type(16))) __bf16 bf16x16;
typedef __attribute__((ext_vector_type(8)))  float  f32x8;

#define DIM 1024
#define NWIT 64
#define B_ 4
#define SQ 2048
#define SK 2048

__device__ __forceinline__ __bf16 f2bf(float f) {
  unsigned u = __float_as_uint(f);
  u += 0x7FFFu + ((u >> 16) & 1u);           // round-to-nearest-even
  unsigned short h = (unsigned short)(u >> 16);
  union { unsigned short s; __bf16 b; } cv;
  cv.s = h;
  return cv.b;
}

template <typename CT> __device__ __forceinline__ CT cvtOut(float f);
template <> __device__ __forceinline__ float  cvtOut<float>(float f)  { return f; }
template <> __device__ __forceinline__ __bf16 cvtOut<__bf16>(float f) { return f2bf(f); }

// Low 32 bits of a generic LDS pointer == wave-relative LDS byte offset
// (ISA 10.2).
__device__ __forceinline__ unsigned lds_off(const void* p) {
  return (unsigned)(unsigned long long)p;
}

// Async-copy 64 contiguous bytes global -> LDS (4x B128, ASYNCcnt += 4).
// INST_OFFSET is added to BOTH the LDS and global addresses (ISA 10.7.4),
// so one address pair covers the whole 64B chunk.
__device__ __forceinline__ void async_copy64(unsigned lds, const void* g) {
  asm volatile(
      "global_load_async_to_lds_b128 %0, %1, off\n\t"
      "global_load_async_to_lds_b128 %0, %1, off offset:16\n\t"
      "global_load_async_to_lds_b128 %0, %1, off offset:32\n\t"
      "global_load_async_to_lds_b128 %0, %1, off offset:48"
      :: "v"(lds), "v"((unsigned long long)g) : "memory");
}

// ---------------------------------------------------------------------------
// Top-64 witness extraction + densify to bf16. One 256-thread block per token.
// ---------------------------------------------------------------------------
__global__ __launch_bounds__(256)
void topk_densify(const float* __restrict__ x, __bf16* __restrict__ xd) {
  const int token = blockIdx.x;
  const float* xp = x + (size_t)token * DIM;
  __bf16* yp = xd + (size_t)token * DIM;
  const int t = threadIdx.x;

  float    v[4];
  unsigned key[4];
#pragma unroll
  for (int i = 0; i < 4; ++i) {
    v[i]   = xp[t + i * 256];
    key[i] = __float_as_uint(fabsf(v[i]));
  }

  __shared__ int red[256];
  unsigned thr = 0u;
  for (int bit = 30; bit >= 0; --bit) {
    const unsigned cand = thr | (1u << bit);
    int c = 0;
#pragma unroll
    for (int i = 0; i < 4; ++i) c += (key[i] >= cand) ? 1 : 0;
    red[t] = c;
    __syncthreads();
    for (int s = 128; s > 0; s >>= 1) {
      if (t < s) red[t] += red[t + s];
      __syncthreads();
    }
    const int total = red[0];
    __syncthreads();
    if (total >= NWIT) thr = cand;
  }

#pragma unroll
  for (int i = 0; i < 4; ++i)
    yp[t + i * 256] = (key[i] >= thr) ? f2bf(v[i]) : f2bf(0.0f);
}

// ---------------------------------------------------------------------------
// f32 -> bf16 elementwise convert.
// ---------------------------------------------------------------------------
__global__ __launch_bounds__(256)
void cvt_f32_bf16(const float* __restrict__ x, __bf16* __restrict__ y, int n) {
  int i = blockIdx.x * 256 + threadIdx.x;
  if (i < n) y[i] = f2bf(x[i]);
}

// ---------------------------------------------------------------------------
// NT GEMM: C[z][M,N] = A[z][M,K] * B[z][N,K]^T, bf16 inputs, f32 accumulate.
// Block = 256 threads = 8 waves; tile 128x128, BK=64; each wave owns 64x32 as
// 4x2 V_WMMA_F32_16X16X32_BF16 accumulators. Double-buffered async staging:
//   iter i: issue stage(i+1) async copies -> s_wait_asynccnt 8 -> barrier ->
//           2x (frag loads + 8 wmma) -> barrier.
// Fragment placement per ISA 7.12.2 (wave32):
//   A 16x32 : lane(0-15)=row M, halves 0..7 = K[sel*8..+7], 8..15 = K[16+sel*8..+7]
//   B 32x16 : lane(0-15)=col N, halves 0..7 = K[sel*8..+7], 8..15 = K[16+sel*8..+7]   (sel = lane>>4)
//   C 16x16 : VGPR r -> M = r + sel*8, N = lane&15
// ---------------------------------------------------------------------------
#define BK 64
#define LDS_STRIDE 72   // 64 + 8 halves pad; rows = 144B (16B-aligned, 36-bank
                        // stride -> conflict-free column reads across 16 lanes)

union FragBF { uint4 u[2]; bf16x16 v; };

template <typename CT, bool TRANSC>
__global__ __launch_bounds__(256)
void gemm_bf16_nt(const __bf16* __restrict__ A, const __bf16* __restrict__ B,
                  CT* __restrict__ C, int M, int N, int K,
                  int lda, int ldb, int ldc,
                  long long strideA, long long strideB, long long strideC) {
  __shared__ __attribute__((aligned(16))) __bf16 As[2][128 * LDS_STRIDE];
  __shared__ __attribute__((aligned(16))) __bf16 Bs[2][128 * LDS_STRIDE];

  A += (size_t)blockIdx.z * strideA;
  B += (size_t)blockIdx.z * strideB;
  C += (size_t)blockIdx.z * strideC;

  const int tileM = blockIdx.y * 128;
  const int tileN = blockIdx.x * 128;

  const int t     = threadIdx.x;
  const int ldRow = t >> 1;          // 0..127
  const int ldCol = (t & 1) * 32;    // 0 or 32 halves (64B chunk)

  const __bf16* aPtr = A + (size_t)(tileM + ldRow) * lda + ldCol;
  const __bf16* bPtr = B + (size_t)(tileN + ldRow) * ldb + ldCol;

  unsigned ldsA[2], ldsB[2];
  ldsA[0] = lds_off(&As[0][ldRow * LDS_STRIDE + ldCol]);
  ldsA[1] = lds_off(&As[1][ldRow * LDS_STRIDE + ldCol]);
  ldsB[0] = lds_off(&Bs[0][ldRow * LDS_STRIDE + ldCol]);
  ldsB[1] = lds_off(&Bs[1][ldRow * LDS_STRIDE + ldCol]);

  const int lane = t & 31;
  const int wave = t >> 5;           // 0..7
  const int wm   = (wave & 1) * 64;  // wave M offset inside tile
  const int wn   = (wave >> 1) * 32; // wave N offset inside tile
  const int lrow = lane & 15;
  const int lsel = lane >> 4;

  f32x8 acc[4][2] = {};

  // Prologue: stage 0 in flight.
  async_copy64(ldsA[0], aPtr);
  async_copy64(ldsB[0], bPtr);

  int stage = 0;
  for (int k0 = 0; k0 < K; k0 += BK) {
    const bool haveNext = (k0 + BK) < K;
    if (haveNext) {
      // Issue next stage before waiting: overlaps HBM latency with WMMA.
      async_copy64(ldsA[stage ^ 1], aPtr + k0 + BK);
      async_copy64(ldsB[stage ^ 1], bPtr + k0 + BK);
      // Async loads retire in order: <=8 outstanding means the 8 copies of
      // the current stage have landed in LDS.
      asm volatile("s_wait_asynccnt 0x8" ::: "memory");
    } else {
      asm volatile("s_wait_asynccnt 0x0" ::: "memory");
    }
    __syncthreads();                 // current stage visible to every wave

    const __bf16* as = As[stage];
    const __bf16* bs = Bs[stage];
#pragma unroll
    for (int kk = 0; kk < BK; kk += 32) {
      FragBF fa[4], fb[2];
#pragma unroll
      for (int mt = 0; mt < 4; ++mt) {
        const int r = wm + mt * 16 + lrow;
        fa[mt].u[0] = *(const uint4*)&as[r * LDS_STRIDE + kk + lsel * 8];
        fa[mt].u[1] = *(const uint4*)&as[r * LDS_STRIDE + kk + 16 + lsel * 8];
      }
#pragma unroll
      for (int nt = 0; nt < 2; ++nt) {
        const int c = wn + nt * 16 + lrow;
        fb[nt].u[0] = *(const uint4*)&bs[c * LDS_STRIDE + kk + lsel * 8];
        fb[nt].u[1] = *(const uint4*)&bs[c * LDS_STRIDE + kk + 16 + lsel * 8];
      }
#pragma unroll
      for (int mt = 0; mt < 4; ++mt)
#pragma unroll
        for (int nt = 0; nt < 2; ++nt)
          acc[mt][nt] = __builtin_amdgcn_wmma_f32_16x16x32_bf16(
              false, fa[mt].v, false, fb[nt].v, (short)0, acc[mt][nt],
              false, false);
    }

    if (haveNext) __syncthreads();   // all reads done before stage^1 refill
    stage ^= 1;
  }

  // ---- store C (optionally transposed) ----
#pragma unroll
  for (int mt = 0; mt < 4; ++mt) {
#pragma unroll
    for (int nt = 0; nt < 2; ++nt) {
      const int col = tileN + wn + nt * 16 + lrow;
#pragma unroll
      for (int r = 0; r < 8; ++r) {
        const int row = tileM + wm + mt * 16 + r + lsel * 8;
        const float val = acc[mt][nt][r];
        if (TRANSC)
          C[(size_t)col * ldc + row] = cvtOut<CT>(val);
        else
          C[(size_t)row * ldc + col] = cvtOut<CT>(val);
      }
    }
  }
}

// ---------------------------------------------------------------------------
// Row softmax over Sk=2048, in place on f32 (d_out attn region) + bf16 copy.
// ---------------------------------------------------------------------------
__global__ __launch_bounds__(256)
void softmax_row(float* __restrict__ attn, __bf16* __restrict__ attn_bf) {
  const int row = blockIdx.x;
  float* p = attn + (size_t)row * SK;
  __bf16* pb = attn_bf + (size_t)row * SK;
  const int t = threadIdx.x;

  float v[8];
  float mx = -3.4e38f;
#pragma unroll
  for (int i = 0; i < 8; ++i) {
    v[i] = p[t + i * 256];
    mx = fmaxf(mx, v[i]);
  }

  __shared__ float red[256];
  red[t] = mx;
  __syncthreads();
  for (int s = 128; s > 0; s >>= 1) {
    if (t < s) red[t] = fmaxf(red[t], red[t + s]);
    __syncthreads();
  }
  mx = red[0];
  __syncthreads();

  float sum = 0.0f;
#pragma unroll
  for (int i = 0; i < 8; ++i) {
    v[i] = __expf(v[i] - mx);
    sum += v[i];
  }
  red[t] = sum;
  __syncthreads();
  for (int s = 128; s > 0; s >>= 1) {
    if (t < s) red[t] += red[t + s];
    __syncthreads();
  }
  const float inv = 1.0f / red[0];
  __syncthreads();

#pragma unroll
  for (int i = 0; i < 8; ++i) {
    const float a = v[i] * inv;
    p[t + i * 256]  = a;
    pb[t + i * 256] = f2bf(a);
  }
}

// ---------------------------------------------------------------------------
// Host-side orchestration.
// ---------------------------------------------------------------------------
extern "C" void kernel_launch(void* const* d_in, const int* in_sizes, int n_in,
                              void* d_out, int out_size, void* d_ws, size_t ws_size,
                              hipStream_t stream) {
  (void)in_sizes; (void)n_in; (void)out_size; (void)ws_size;

  const float* q  = (const float*)d_in[0];
  const float* k  = (const float*)d_in[1];
  const float* v  = (const float*)d_in[2];
  const float* Wv = (const float*)d_in[3];
  const float* Wo = (const float*)d_in[4];

  const size_t nTok   = (size_t)B_ * SQ;                        // 8192 tokens
  const size_t tokBF  = nTok * DIM * sizeof(__bf16);            // 16.78 MB
  const size_t wBF    = (size_t)DIM * DIM * sizeof(__bf16);     // 2.10 MB
  const size_t attnBF = (size_t)B_ * SQ * SK * sizeof(__bf16);  // 33.55 MB

  char* ws = (char*)d_ws;
  size_t off = 0;
  auto bump = [&](size_t sz) { size_t o = off; off = (off + sz + 255) & ~(size_t)255; return o; };
  __bf16* qd     = (__bf16*)(ws + bump(tokBF));
  __bf16* kd     = (__bf16*)(ws + bump(tokBF));
  __bf16* vb     = (__bf16*)(ws + bump(tokBF));
  __bf16* Wvb    = (__bf16*)(ws + bump(wBF));
  __bf16* Wob    = (__bf16*)(ws + bump(wBF));
  __bf16* vprojT = (__bf16*)(ws + bump(tokBF));   // per batch: (D, Sk)
  __bf16* attnbf = (__bf16*)(ws + bump(attnBF));
  __bf16* out1   = (__bf16*)(ws + bump(tokBF));   // ~122 MB total

  float* out  = (float*)d_out;                        // (B, Sq, D)
  float* attn = out + (size_t)B_ * SQ * DIM;          // (B, Sq, Sk)

  // 1) witness extraction + densify
  topk_densify<<<dim3((unsigned)nTok), 256, 0, stream>>>(q, qd);
  topk_densify<<<dim3((unsigned)nTok), 256, 0, stream>>>(k, kd);

  // 2) bf16 conversions
  cvt_f32_bf16<<<dim3((B_ * SK * DIM + 255) / 256), 256, 0, stream>>>(v, vb, B_ * SK * DIM);
  cvt_f32_bf16<<<dim3((DIM * DIM + 255) / 256), 256, 0, stream>>>(Wv, Wvb, DIM * DIM);
  cvt_f32_bf16<<<dim3((DIM * DIM + 255) / 256), 256, 0, stream>>>(Wo, Wob, DIM * DIM);

  // 3) scores = qd @ kd^T  (f32 directly into d_out attn region; TAU = 1)
  gemm_bf16_nt<float, false><<<dim3(SK / 128, SQ / 128, B_), 256, 0, stream>>>(
      qd, kd, attn, SQ, SK, DIM, DIM, DIM, SK,
      (long long)SQ * DIM, (long long)SK * DIM, (long long)SQ * SK);

  // 4) vprojT = (v @ Wv^T)^T  (stored as (D, Sk) per batch so GEMM-6 is NT)
  gemm_bf16_nt<__bf16, true><<<dim3(DIM / 128, SK / 128, B_), 256, 0, stream>>>(
      vb, Wvb, vprojT, SK, DIM, DIM, DIM, DIM, SK,
      (long long)SK * DIM, 0LL, (long long)DIM * SK);

  // 5) softmax (in place on d_out attn) + bf16 copy
  softmax_row<<<dim3((unsigned)(B_ * SQ)), 256, 0, stream>>>(attn, attnbf);

  // 6) out1 = attn_bf @ vprojT^T
  gemm_bf16_nt<__bf16, false><<<dim3(DIM / 128, SQ / 128, B_), 256, 0, stream>>>(
      attnbf, vprojT, out1, SQ, DIM, SK, SK, SK, DIM,
      (long long)SQ * SK, (long long)DIM * SK, (long long)SQ * DIM);

  // 7) out = out1 @ Wo^T  (f32 into d_out)
  gemm_bf16_nt<float, false><<<dim3(DIM / 128, SQ / 128, B_), 256, 0, stream>>>(
      out1, Wob, out, SQ, DIM, DIM, DIM, DIM, DIM,
      (long long)SQ * DIM, 0LL, (long long)SQ * DIM);
}
